// RecentResidualBank_13950053777866
// MI455X (gfx1250) — matrix operand order repaired and checked by reading, not compiled, not run on hardware
//
#include <hip/hip_runtime.h>
#include <math.h>

typedef __attribute__((ext_vector_type(16))) _Float16 v16h;
typedef __attribute__((ext_vector_type(4)))  _Float16 v4h;
typedef __attribute__((ext_vector_type(8)))  float    v8f;
typedef __attribute__((ext_vector_type(4)))  float    v4f;

#define NN    325
#define GTOT  2016      // 7 * 288
#define RT_PER_G 21     // ceil(325/16)
#define TILES_TOTAL (GTOT * RT_PER_G)   // 42336

__device__ inline v8f wmma32(v16h a, v16h b, v8f c) {
  return __builtin_amdgcn_wmma_f32_16x16x32_f16(false, a, false, b, (short)0, c, false, false);
}

__device__ inline float sigmoidf_(float x) { return 1.f / (1.f + __expf(-x)); }

__device__ inline void atomicMaxF(float* addr, float v) {
  int* ia = (int*)addr;
  int cur = __float_as_int(*addr);
  while (__int_as_float(cur) < v) {
    int prev = atomicCAS(ia, cur, __float_as_int(v));
    if (prev == cur) break;
    cur = prev;
  }
}

// ---------------------------------------------------------------------------
// Kernel 1: GRU (input 1 -> hidden 32, K=12 steps) + recent-proj MLP 32->32->32
// One wave32 per node, lane = hidden unit.
// ---------------------------------------------------------------------------
__global__ void __launch_bounds__(256)
gru_recent_kernel(const float* __restrict__ seq,
                  const float* __restrict__ w_ih, const float* __restrict__ w_hh,
                  const float* __restrict__ b_ih, const float* __restrict__ b_hh,
                  const float* __restrict__ rw1, const float* __restrict__ rb1,
                  const float* __restrict__ rw2, const float* __restrict__ rb2,
                  float* __restrict__ outr)
{
  __shared__ float whh[32*96];
  __shared__ float w1[32*32];
  __shared__ float w2[32*32];
  const int tid = threadIdx.x;
  for (int i = tid; i < 32*96; i += 256) whh[i] = w_hh[i];
  for (int i = tid; i < 1024;  i += 256) { w1[i] = rw1[i]; w2[i] = rw2[i]; }
  __syncthreads();

  const int lane = tid & 31;
  const int n = blockIdx.x * 8 + (tid >> 5);
  if (n >= NN) return;

  const float wr = w_ih[lane], wz = w_ih[32+lane], wn = w_ih[64+lane];
  const float br = b_ih[lane], bz = b_ih[32+lane], bn = b_ih[64+lane];
  const float hbr = b_hh[lane], hbz = b_hh[32+lane], hbn = b_hh[64+lane];
  float h = 0.f;
  for (int t = 0; t < 12; ++t) {
    const float x = seq[n*12 + t];
    float hr = hbr, hz = hbz, hn = hbn;
    #pragma unroll 8
    for (int k = 0; k < 32; ++k) {
      const float hk = __shfl(h, k);
      hr += hk * whh[k*96 + lane];
      hz += hk * whh[k*96 + 32 + lane];
      hn += hk * whh[k*96 + 64 + lane];
    }
    const float r  = sigmoidf_(x*wr + br + hr);
    const float z  = sigmoidf_(x*wz + bz + hz);
    const float nn = tanhf(x*wn + bn + r*hn);
    h = (1.f - z) * nn + z * h;
  }
  float acc = rb1[lane];
  #pragma unroll 8
  for (int k = 0; k < 32; ++k) acc += __shfl(h, k) * w1[k*32 + lane];
  const float h1 = acc > 0.f ? acc : 0.f;
  float acc2 = rb2[lane];
  #pragma unroll 8
  for (int k = 0; k < 32; ++k) acc2 += __shfl(h1, k) * w2[k*32 + lane];
  outr[n*32 + lane] = acc2;
}

// ---------------------------------------------------------------------------
// Kernel 2: calendar encoder, one wave per (weekday, slot) cell, lane = unit.
// ---------------------------------------------------------------------------
__global__ void __launch_bounds__(256)
calendar_kernel(const float* __restrict__ wde, const float* __restrict__ sle,
                const float* __restrict__ cw1, const float* __restrict__ cb1,
                const float* __restrict__ cw2, const float* __restrict__ cb2,
                float* __restrict__ outc)
{
  __shared__ float w1[29*32];
  __shared__ float w2[32*32];
  const int tid = threadIdx.x;
  for (int i = tid; i < 29*32; i += 256) w1[i] = cw1[i];
  for (int i = tid; i < 1024;  i += 256) w2[i] = cw2[i];
  __syncthreads();

  const int lane = tid & 31;
  const int g = blockIdx.x * 8 + (tid >> 5);
  if (g >= GTOT) return;
  const int wd = g / 288, sl = g - wd*288;

  float f[29];
  #pragma unroll
  for (int i = 0; i < 8;  ++i) f[i]   = wde[wd*8 + i];
  #pragma unroll
  for (int i = 0; i < 16; ++i) f[8+i] = sle[sl*16 + i];
  const float ta = 6.283185307179586f * (float)sl / 288.f;
  const float wa = 6.283185307179586f * (float)wd / 7.f;
  f[24] = sinf(ta); f[25] = cosf(ta); f[26] = sinf(wa); f[27] = cosf(wa);
  f[28] = (wd >= 5) ? 1.f : 0.f;

  float acc = cb1[lane];
  #pragma unroll
  for (int i = 0; i < 29; ++i) acc += f[i] * w1[i*32 + lane];
  const float h1 = acc > 0.f ? acc : 0.f;
  float acc2 = cb2[lane];
  #pragma unroll 8
  for (int k = 0; k < 32; ++k) acc2 += __shfl(h1, k) * w2[k*32 + lane];
  outc[g*32 + lane] = acc2;
}

// ---------------------------------------------------------------------------
// Kernel 3: fusion + delta MLP (128->64 relu ->64) + sigmoid gate (128->64).
// WMMA f32_16x16x32_f16, one wave per 16-row tile. Weights are staged in LDS
// pre-swizzled into WMMA B-fragment order: fragment f, lane l -> 16 contiguous
// halves at (f*32+l)*16, so each B fragment is a single 32B LDS vector load.
// Dynamic LDS (halves): dm_w1 8192 | gm_w 8192 | dm_w2 4096 | 8*2048 staging.
// ---------------------------------------------------------------------------
__global__ void __launch_bounds__(256)
fusion_delta_kernel(const float* __restrict__ Hb,
                    const float* __restrict__ rec,
                    const float* __restrict__ cal,
                    const float* __restrict__ dm_w1, const float* __restrict__ dm_b1,
                    const float* __restrict__ dm_w2, const float* __restrict__ dm_b2,
                    const float* __restrict__ gm_w,  const float* __restrict__ gm_b,
                    float* __restrict__ out)
{
  extern __shared__ _Float16 sh[];
  _Float16* w1s = sh;              // 16 frags * 512  (dm_w1, K=128)
  _Float16* gws = sh + 8192;       // 16 frags * 512  (gm_w,  K=128)
  _Float16* w2s = sh + 16384;      // 8  frags * 512  (dm_w2, K=64)
  const int tid  = threadIdx.x;
  const int lane = tid & 31;
  const int wv   = tid >> 5;
  _Float16* stg = sh + 20480 + wv * 2048;   // per-wave 16x128 staging

  // swizzle weights into fragment order
  for (int t = tid; t < 512; t += 256) {           // frags 0..15 (kc*4+nt)
    const int f = t >> 5, ln = t & 31;
    const int kc = f >> 2, nt = f & 3;
    const int kbase = kc*32 + ((ln >> 4) << 3);
    const int n = nt*16 + (ln & 15);
    v16h bv, gv;
    #pragma unroll
    for (int e = 0; e < 16; ++e) {
      bv[e] = (_Float16)dm_w1[(kbase + e + ((e >> 3) << 3))*64 + n];
      gv[e] = (_Float16)gm_w [(kbase + e + ((e >> 3) << 3))*64 + n];
    }
    *(v16h*)(w1s + t*16) = bv;
    *(v16h*)(gws + t*16) = gv;
  }
  for (int t = tid; t < 256; t += 256) {           // frags 0..7 (kc*4+nt, kc<2)
    const int f = t >> 5, ln = t & 31;
    const int kc = f >> 2, nt = f & 3;
    const int kbase = kc*32 + ((ln >> 4) << 3);
    const int n = nt*16 + (ln & 15);
    v16h bv;
    #pragma unroll
    for (int e = 0; e < 16; ++e) bv[e] = (_Float16)dm_w2[(kbase + e + ((e >> 3) << 3))*64 + n];
    *(v16h*)(w2s + t*16) = bv;
  }
  __syncthreads();

  const int tile = blockIdx.x * 8 + wv;     // grid sized so tile < TILES_TOTAL
  const int g  = tile / RT_PER_G;
  const int rt = tile - g * RT_PER_G;
  const int n0 = rt * 16;

  // stage fusion rows [16 x 128] = [H_base(64) | rec(32) | cal(32)] as f16
  for (int i = lane*4; i < 2048; i += 128) {
    const int r = i >> 7, c = i & 127;
    int n = n0 + r; if (n > NN-1) n = NN-1;
    v4f v;
    if      (c < 64) v = *(const v4f*)(Hb + (g*NN + n)*64 + c);
    else if (c < 96) v = *(const v4f*)(rec + n*32 + (c - 64));
    else             v = *(const v4f*)(cal + g*32 + (c - 96));
    v4h hv; hv[0] = (_Float16)v[0]; hv[1] = (_Float16)v[1];
            hv[2] = (_Float16)v[2]; hv[3] = (_Float16)v[3];
    *(v4h*)(stg + i) = hv;
  }
  __syncthreads();

  const int Mh     = lane & 15;
  const int hi8    = (lane >> 4) << 3;   // A half-K / C row offset
  const int fb     = lane * 16;          // B-fragment lane offset

  // A fragments (16x32 f16 layout), K = 128 -> 4 chunks
  v16h a[4];
  #pragma unroll
  for (int kc = 0; kc < 4; ++kc) {
    const _Float16* p = stg + Mh*128 + kc*32 + hi8;
    #pragma unroll
    for (int e = 0; e < 8; ++e) { a[kc][e] = p[e]; a[kc][8+e] = p[16+e]; }
  }

  // layer1 + gate
  float gacc[4][8];
  #pragma unroll
  for (int nt = 0; nt < 4; ++nt) {
    v8f acc = {}; v8f accg = {};
    const int nb = nt*16 + Mh;
    #pragma unroll
    for (int kc = 0; kc < 4; ++kc) {
      const v16h b  = *(const v16h*)(w1s + (kc*4 + nt)*512 + fb);
      const v16h bg = *(const v16h*)(gws + (kc*4 + nt)*512 + fb);
      acc  = wmma32(a[kc], b,  acc);
      accg = wmma32(a[kc], bg, accg);
    }
    const float b1v = dm_b1[nb];
    #pragma unroll
    for (int r = 0; r < 8; ++r) {
      float h1 = acc[r] + b1v; h1 = h1 > 0.f ? h1 : 0.f;
      stg[(r + hi8)*64 + nb] = (_Float16)h1;   // reuse staging as h1[16][64]
      gacc[nt][r] = accg[r];
    }
  }
  __syncthreads();

  // layer2: h1(16x64) @ dm_w2(64x64), then gate
  v16h a2[2];
  #pragma unroll
  for (int kc = 0; kc < 2; ++kc) {
    const _Float16* p = stg + Mh*64 + kc*32 + hi8;
    #pragma unroll
    for (int e = 0; e < 8; ++e) { a2[kc][e] = p[e]; a2[kc][8+e] = p[16+e]; }
  }
  #pragma unroll
  for (int nt = 0; nt < 4; ++nt) {
    v8f acc = {};
    const int nb = nt*16 + Mh;
    #pragma unroll
    for (int kc = 0; kc < 2; ++kc) {
      const v16h b = *(const v16h*)(w2s + (kc*4 + nt)*512 + fb);
      acc = wmma32(a2[kc], b, acc);
    }
    const float b2v = dm_b2[nb], gbv = gm_b[nb];
    #pragma unroll
    for (int r = 0; r < 8; ++r) {
      const int n = n0 + r + hi8;
      if (n < NN) {
        const float gate = sigmoidf_(gacc[nt][r] + gbv);
        out[(g*NN + n)*64 + nb] = (acc[r] + b2v) * gate;
      }
    }
  }
}

// ---------------------------------------------------------------------------
// Kernel 4: GATConv (self loops appended, concat=False -> head mean), one
// workgroup per grid cell g. WMMA projection (swizzled weight frags in LDS),
// LDS segmented softmax + scatter.
// Dynamic LDS: x f16[336*64] | wfrag f16[2*H*4*512] | xp f16[336*H*64] |
//              out f32[325*64] | a_s,a_d,segmax,segsum f32[325*H]
// ---------------------------------------------------------------------------
__global__ void __launch_bounds__(256)
gat_kernel(const float* __restrict__ xin, const int* __restrict__ ei, int E,
           const float* __restrict__ W, const float* __restrict__ att_s,
           const float* __restrict__ att_d, const float* __restrict__ bias,
           int H, float* __restrict__ xout)
{
  extern __shared__ char smemraw[];
  const int ldxp   = H * 64;
  const int ntiles = H * 4;
  _Float16* xs  = (_Float16*)smemraw;            // 336*64
  _Float16* wfs = xs + 336*64;                   // 2*ntiles*512
  _Float16* xps = wfs + 2*ntiles*512;            // 336*ldxp
  float* outm = (float*)(xps + 336*ldxp);        // 325*64
  float* as   = outm + NN*64;
  float* ad   = as + NN*H;
  float* smax = ad + NN*H;
  float* ssum = smax + NN*H;

  const int g    = blockIdx.x;
  const int tid  = threadIdx.x;
  const int lane = tid & 31;
  const int wv   = tid >> 5;
  const int nE   = E + NN;

  // stage x (f32 -> f16, 4-wide), zero-pad rows 325..335
  for (int i = tid*4; i < 336*64; i += 1024) {
    const int r = i >> 6;
    v4h hv;
    if (r < NN) {
      const v4f v = *(const v4f*)(xin + (g*NN + r)*64 + (i & 63));
      hv[0] = (_Float16)v[0]; hv[1] = (_Float16)v[1];
      hv[2] = (_Float16)v[2]; hv[3] = (_Float16)v[3];
    } else {
      hv[0] = hv[1] = hv[2] = hv[3] = (_Float16)0.f;
    }
    *(v4h*)(xs + i) = hv;
  }
  // stage W swizzled into fragment order (frag f = kc*ntiles + nt)
  for (int t = tid; t < 2*ntiles*32; t += 256) {
    const int f = t >> 5, ln = t & 31;
    const int kc = f / ntiles, nt = f - kc*ntiles;
    const int kbase = kc*32 + ((ln >> 4) << 3);
    const int n = nt*16 + (ln & 15);
    v16h bv;
    #pragma unroll
    for (int e = 0; e < 16; ++e) bv[e] = (_Float16)W[(kbase + e + ((e >> 3) << 3))*ldxp + n];
    *(v16h*)(wfs + t*16) = bv;
  }
  __syncthreads();

  // xp = x @ W  (325x64 @ 64x(H*64)) via WMMA
  const int Mh  = lane & 15;
  const int hi8 = (lane >> 4) << 3;
  const int fb  = lane * 16;
  for (int t = wv; t < RT_PER_G * ntiles; t += 8) {
    const int rt = t / ntiles, nt = t - rt*ntiles;
    const int nb = nt*16 + Mh;
    v16h a0, a1;
    const _Float16* p0 = xs + (rt*16 + Mh)*64 + hi8;
    #pragma unroll
    for (int e = 0; e < 8; ++e) {
      a0[e] = p0[e];      a0[8+e] = p0[16+e];
      a1[e] = p0[32+e];   a1[8+e] = p0[48+e];
    }
    const v16h b0 = *(const v16h*)(wfs + (0*ntiles + nt)*512 + fb);
    const v16h b1 = *(const v16h*)(wfs + (1*ntiles + nt)*512 + fb);
    v8f acc = {};
    acc = wmma32(a0, b0, acc);
    acc = wmma32(a1, b1, acc);
    #pragma unroll
    for (int r = 0; r < 8; ++r)
      xps[(rt*16 + r + hi8)*ldxp + nb] = (_Float16)acc[r];
  }
  __syncthreads();

  // attention scalars + init
  for (int i = tid; i < NN*H; i += 256) {
    const int n = i / H, h = i - n*H;
    float s1 = 0.f, s2 = 0.f;
    for (int c = 0; c < 64; ++c) {
      const float xv = (float)xps[n*ldxp + h*64 + c];
      s1 += xv * att_s[h*64 + c];
      s2 += xv * att_d[h*64 + c];
    }
    as[i] = s1; ad[i] = s2; smax[i] = -1e30f; ssum[i] = 0.f;
  }
  for (int i = tid; i < NN*64; i += 256) outm[i] = 0.f;
  __syncthreads();

  // pass 1: segment max per dst
  for (int e = tid; e < nE; e += 256) {
    int s, d;
    if (e < E) { s = ei[e]; d = ei[E + e]; } else { s = d = e - E; }
    for (int h = 0; h < H; ++h) {
      float v = as[s*H + h] + ad[d*H + h];
      v = v > 0.f ? v : 0.2f * v;
      atomicMaxF(&smax[d*H + h], v);
    }
  }
  __syncthreads();

  // pass 2: segment sum of exp
  for (int e = tid; e < nE; e += 256) {
    int s, d;
    if (e < E) { s = ei[e]; d = ei[E + e]; } else { s = d = e - E; }
    for (int h = 0; h < H; ++h) {
      float v = as[s*H + h] + ad[d*H + h];
      v = v > 0.f ? v : 0.2f * v;
      atomicAdd(&ssum[d*H + h], __expf(v - smax[d*H + h]));
    }
  }
  __syncthreads();

  // pass 3: aggregate alpha * xp[src] into out (mean over heads folded in)
  const float invH = 1.f / (float)H;
  for (int e = tid; e < nE; e += 256) {
    int s, d;
    if (e < E) { s = ei[e]; d = ei[E + e]; } else { s = d = e - E; }
    for (int h = 0; h < H; ++h) {
      float v = as[s*H + h] + ad[d*H + h];
      v = v > 0.f ? v : 0.2f * v;
      const float alpha = __expf(v - smax[d*H + h]) / (ssum[d*H + h] + 1e-16f) * invH;
      const _Float16* xr = xps + s*ldxp + h*64;
      for (int c = 0; c < 64; ++c)
        atomicAdd(&outm[d*64 + c], alpha * (float)xr[c]);
    }
  }
  __syncthreads();

  // write relu(out + bias)
  for (int i = tid; i < NN*64; i += 256) {
    const float v = outm[i] + bias[i & 63];
    xout[g*NN*64 + i] = v > 0.f ? v : 0.f;
  }
}

// ---------------------------------------------------------------------------
// Kernel 5: finalize. hadBuf holds delta_final; copy into deltaOut, form
// H_adapted = H_base + delta in place, and run speed head 64->32->1 per row.
// One wave per row, lane = hidden unit (32) / 2 feature elems.
// ---------------------------------------------------------------------------
__global__ void __launch_bounds__(256)
finalize_kernel(const float* __restrict__ Hb,
                float* __restrict__ deltaOut,
                float* __restrict__ hadBuf,
                const float* __restrict__ sw1, const float* __restrict__ sb1,
                const float* __restrict__ sw2, const float* __restrict__ sb2,
                float* __restrict__ pred)
{
  __shared__ float w1[64*32];
  const int tid = threadIdx.x;
  for (int i = tid; i < 2048; i += 256) w1[i] = sw1[i];
  __syncthreads();

  const int lane = tid & 31;
  const int wvg  = (blockIdx.x * 256 + tid) >> 5;
  const int nw   = (gridDim.x * 256) >> 5;
  const float sw2l = sw2[lane];
  const float sb1l = sb1[lane];
  const float sb2v = sb2[0];

  for (int row = wvg; row < GTOT*NN; row += nw) {
    const int base = row * 64;
    const float d0 = hadBuf[base + 2*lane];
    const float d1 = hadBuf[base + 2*lane + 1];
    deltaOut[base + 2*lane]     = d0;
    deltaOut[base + 2*lane + 1] = d1;
    const float had0 = Hb[base + 2*lane]     + d0;
    const float had1 = Hb[base + 2*lane + 1] + d1;
    hadBuf[base + 2*lane]     = had0;
    hadBuf[base + 2*lane + 1] = had1;

    float acc = sb1l;
    #pragma unroll 8
    for (int k = 0; k < 32; ++k) {
      const float a0 = __shfl(had0, k);
      const float a1 = __shfl(had1, k);
      acc += a0 * w1[(2*k)*32 + lane] + a1 * w1[(2*k + 1)*32 + lane];
    }
    const float h1 = acc > 0.f ? acc : 0.f;
    float p = h1 * sw2l;
    #pragma unroll
    for (int off = 16; off > 0; off >>= 1) p += __shfl_xor(p, off);
    if (lane == 0) pred[row] = p + sb2v;
  }
}

// ---------------------------------------------------------------------------
extern "C" void kernel_launch(void* const* d_in, const int* in_sizes, int n_in,
                              void* d_out, int out_size, void* d_ws, size_t ws_size,
                              hipStream_t stream) {
  const float* Hb     = (const float*)d_in[0];
  const float* seq    = (const float*)d_in[1];
  const int*   ei     = (const int*)d_in[2];
  const float* gw_ih  = (const float*)d_in[3];
  const float* gw_hh  = (const float*)d_in[4];
  const float* gb_ih  = (const float*)d_in[5];
  const float* gb_hh  = (const float*)d_in[6];
  const float* rp_w1  = (const float*)d_in[7];
  const float* rp_b1  = (const float*)d_in[8];
  const float* rp_w2  = (const float*)d_in[9];
  const float* rp_b2  = (const float*)d_in[10];
  const float* wde    = (const float*)d_in[11];
  const float* sle    = (const float*)d_in[12];
  const float* cw1    = (const float*)d_in[13];
  const float* cb1    = (const float*)d_in[14];
  const float* cw2    = (const float*)d_in[15];
  const float* cb2    = (const float*)d_in[16];
  const float* dm_w1  = (const float*)d_in[17];
  const float* dm_b1  = (const float*)d_in[18];
  const float* dm_w2  = (const float*)d_in[19];
  const float* dm_b2  = (const float*)d_in[20];
  const float* gm_w   = (const float*)d_in[21];
  const float* gm_b   = (const float*)d_in[22];
  const float* g1_w   = (const float*)d_in[23];
  const float* g1_as  = (const float*)d_in[24];
  const float* g1_ad  = (const float*)d_in[25];
  const float* g1_b   = (const float*)d_in[26];
  const float* g2_w   = (const float*)d_in[27];
  const float* g2_as  = (const float*)d_in[28];
  const float* g2_ad  = (const float*)d_in[29];
  const float* g2_b   = (const float*)d_in[30];
  const float* sh_w1  = (const float*)d_in[31];
  const float* sh_b1  = (const float*)d_in[32];
  const float* sh_w2  = (const float*)d_in[33];
  const float* sh_b2  = (const float*)d_in[34];

  const int E = in_sizes[2] / 2;

  float* ws_rec = (float*)d_ws;          // 325*32
  float* ws_cal = ws_rec + NN*32;        // 2016*32

  float* outDelta = (float*)d_out;                          // [G,N,64]
  float* outHad   = outDelta + (size_t)GTOT*NN*64;          // [G,N,64]
  float* outPred  = outHad   + (size_t)GTOT*NN*64;          // [G,N]

  gru_recent_kernel<<<(NN + 7) / 8, 256, 0, stream>>>(
      seq, gw_ih, gw_hh, gb_ih, gb_hh, rp_w1, rp_b1, rp_w2, rp_b2, ws_rec);

  calendar_kernel<<<GTOT / 8, 256, 0, stream>>>(
      wde, sle, cw1, cb1, cw2, cb2, ws_cal);

  // fusion + delta MLP + gate  ->  pre-GAT delta into H_adapted slot
  const size_t smFusion = (size_t)(20480 + 8 * 2048) * sizeof(_Float16);  // 73728 B
  fusion_delta_kernel<<<TILES_TOTAL / 8, 256, smFusion, stream>>>(
      Hb, ws_rec, ws_cal, dm_w1, dm_b1, dm_w2, dm_b2, gm_w, gm_b, outHad);

  auto smGat = [](int H) {
    return (size_t)336*64*2            // x f16
         + (size_t)2*H*4*512*2         // swizzled weight frags f16
         + (size_t)336*H*64*2          // xp f16
         + (size_t)NN*64*4             // out accum f32
         + (size_t)4*NN*H*4;           // a_s, a_d, segmax, segsum f32
  };
  // GAT1 (H=2): read pre-delta (outHad), write x1 -> delta slot
  gat_kernel<<<GTOT, 256, smGat(2), stream>>>(
      outHad, ei, E, g1_w, g1_as, g1_ad, g1_b, 2, outDelta);
  // GAT2 (H=1): read x1 (outDelta), write delta_final -> H_adapted slot
  gat_kernel<<<GTOT, 256, smGat(1), stream>>>(
      outDelta, ei, E, g2_w, g2_as, g2_ad, g2_b, 1, outHad);

  // finalize: delta copy, H_adapted = H_base + delta, speed head
  finalize_kernel<<<1024, 256, 0, stream>>>(
      Hb, outDelta, outHad, sh_w1, sh_b1, sh_w2, sh_b2, outPred);
}
